// MolecularPropertyPredictor_74938589381411
// MI455X (gfx1250) — compile-verified, matrix operand order checked by reading, not run on hardware
//
#include <hip/hip_runtime.h>
#include <hip/hip_bf16.h>
#include <stdint.h>

#define N_NODES 50000
#define N_EDGES 400000
#define N_ETOT  (N_EDGES + N_NODES)   // edges + self loops
#define N_GRAPHS 256
#define F_IN 128
#define HID 64
#define GATW 256                      // HEADS*HID

#define CDIV(a,b) (((a)+(b)-1)/(b))

typedef __attribute__((ext_vector_type(16))) _Float16 v16h;
typedef __attribute__((ext_vector_type(8)))  _Float16 v8h;
typedef __attribute__((ext_vector_type(8)))  float    v8f;

// ---------------- helpers ----------------

static __device__ __forceinline__ void atomicMaxFloat(float* addr, float val) {
    // classic ordered-bits trick: works with -inf init (0xFF800000)
    if (val >= 0.0f) atomicMax((int*)addr, __float_as_int(val));
    else             atomicMin((unsigned int*)addr, __float_as_uint(val));
}

__global__ void fill_f32(float* __restrict__ p, float v, int n) {
    int i = blockIdx.x * blockDim.x + threadIdx.x;
    if (i < n) p[i] = v;
}

__global__ void cvt_f16(const float* __restrict__ s, _Float16* __restrict__ d, int n) {
    int i = blockIdx.x * blockDim.x + threadIdx.x;
    if (i < n) d[i] = (_Float16)s[i];
}

// w[K][N] (row major) -> wT[N][K] f16
__global__ void cvt_T_f16(const float* __restrict__ w, _Float16* __restrict__ wT, int K, int N) {
    int i = blockIdx.x * blockDim.x + threadIdx.x;
    if (i >= K * N) return;
    int k = i / N, n = i - k * N;
    wT[(size_t)n * K + k] = (_Float16)w[i];
}

// ---------------- WMMA GEMM: C[M,N] = A[M,K] * BT[N,K]^T ----------------
// One wave (32 lanes) per 16x16 output tile. K multiple of 32, M,N multiples of 16.
__global__ void gemm_wmma_f16(const _Float16* __restrict__ A,
                              const _Float16* __restrict__ BT,
                              float* __restrict__ C, int M, int N, int K) {
    int tilesN = N >> 4;
    int tile = blockIdx.x;
    int tm = tile / tilesN;
    int tn = tile - tm * tilesN;
    int lane  = threadIdx.x & 31;
    int r     = lane & 15;        // row (A) / col (B) within tile
    int khalf = lane >> 4;        // which K-half this lane holds

    const _Float16* arow = A  + (size_t)(tm * 16 + r) * K;
    const _Float16* brow = BT + (size_t)(tn * 16 + r) * K;

    v8f acc = {0.f, 0.f, 0.f, 0.f, 0.f, 0.f, 0.f, 0.f};
    union V16 { v16h v; v8h h[2]; };

    for (int k0 = 0; k0 < K; k0 += 32) {
        V16 a, b;
        // A-frag layout (16-bit, 16x32): VGPR0-3 hold K = khalf*8 .. +7 ;
        // VGPR4-7 hold K = 16 + khalf*8 .. +7  (two contiguous b128 loads)
        a.h[0] = *(const v8h*)(arow + k0 + khalf * 8);
        a.h[1] = *(const v8h*)(arow + k0 + 16 + khalf * 8);
        b.h[0] = *(const v8h*)(brow + k0 + khalf * 8);
        b.h[1] = *(const v8h*)(brow + k0 + 16 + khalf * 8);
        acc = __builtin_amdgcn_wmma_f32_16x16x32_f16(
            /*neg_a=*/false, a.v, /*neg_b=*/false, b.v,
            /*c_mod=*/(short)0, acc, /*reuse_a=*/false, /*reuse_b=*/false);
    }
    // C/D layout: VGPR i -> M = i + khalf*8, N = lane&15
    float* cp = C + (size_t)(tm * 16 + khalf * 8) * N + tn * 16 + r;
#pragma unroll
    for (int i = 0; i < 8; ++i) cp[(size_t)i * N] = acc[i];
}

// ---------------- graph kernels ----------------

__global__ void deg_kernel(const int* __restrict__ dstv, float* __restrict__ deg) {
    int e = blockIdx.x * blockDim.x + threadIdx.x;
    if (e >= N_ETOT) return;
    int d = (e < N_EDGES) ? dstv[e] : (e - N_EDGES);
    atomicAdd(&deg[d], 1.0f);
}

__global__ void dinv_kernel(const float* __restrict__ deg, float* __restrict__ dinv) {
    int i = blockIdx.x * blockDim.x + threadIdx.x;
    if (i >= N_NODES) return;
    float dg = deg[i];
    dinv[i] = dg > 0.0f ? rsqrtf(dg) : 0.0f;
}

// AGG[dst] += norm * H[src]; 16 lanes per edge, float4 per lane (HID=64)
__global__ void gcn_scatter(const int* __restrict__ srcv, const int* __restrict__ dstv,
                            const float* __restrict__ dinv,
                            const float* __restrict__ H, float* __restrict__ AGG) {
    unsigned t = blockIdx.x * blockDim.x + threadIdx.x;
    unsigned e = t >> 4;
    if (e >= N_ETOT) return;
    int cg = (int)(t & 15u) << 2;
    int s = (e < N_EDGES) ? srcv[e] : (int)(e - N_EDGES);
    int d = (e < N_EDGES) ? dstv[e] : (int)(e - N_EDGES);
    float nrm = dinv[s] * dinv[d];
    const float4 hv = *(const float4*)(H + (size_t)s * HID + cg);
    float* o = AGG + (size_t)d * HID + cg;
    atomicAdd(o + 0, nrm * hv.x);
    atomicAdd(o + 1, nrm * hv.y);
    atomicAdd(o + 2, nrm * hv.z);
    atomicAdd(o + 3, nrm * hv.w);
}

// X[i] = relu(AGG[i] + b[i mod 64])
__global__ void bias_relu(const float* __restrict__ AGG, const float* __restrict__ b,
                          float* __restrict__ X, int n) {
    int i = blockIdx.x * blockDim.x + threadIdx.x;
    if (i < n) X[i] = fmaxf(AGG[i] + b[i & (HID - 1)], 0.0f);
}

// per-(node,head) attention logits: asrc/adst [N,4]
__global__ void gat_scores(const float* __restrict__ G,
                           const float* __restrict__ att_src, const float* __restrict__ att_dst,
                           float* __restrict__ asrc, float* __restrict__ adst) {
    int t = blockIdx.x * blockDim.x + threadIdx.x;
    if (t >= N_NODES * 4) return;
    int nidx = t >> 2, h = t & 3;
    const float* g  = G + (size_t)nidx * GATW + h * HID;
    const float* as = att_src + h * HID;
    const float* ad = att_dst + h * HID;
    float s1 = 0.f, s2 = 0.f;
    for (int c = 0; c < HID; ++c) { float v = g[c]; s1 += v * as[c]; s2 += v * ad[c]; }
    asrc[t] = s1;
    adst[t] = s2;
}

static __device__ __forceinline__ float lrelu02(float x) { return x > 0.f ? x : 0.2f * x; }

__global__ void gat_edge_max(const int* __restrict__ srcv, const int* __restrict__ dstv,
                             const float* __restrict__ asrc, const float* __restrict__ adst,
                             float* __restrict__ m) {
    int t = blockIdx.x * blockDim.x + threadIdx.x;
    if (t >= N_ETOT * 4) return;
    int e = t >> 2, h = t & 3;
    int s = (e < N_EDGES) ? srcv[e] : (e - N_EDGES);
    int d = (e < N_EDGES) ? dstv[e] : (e - N_EDGES);
    float ee = lrelu02(asrc[s * 4 + h] + adst[d * 4 + h]);
    atomicMaxFloat(&m[d * 4 + h], ee);
}

__global__ void gat_edge_denom(const int* __restrict__ srcv, const int* __restrict__ dstv,
                               const float* __restrict__ asrc, const float* __restrict__ adst,
                               const float* __restrict__ m, float* __restrict__ dnm) {
    int t = blockIdx.x * blockDim.x + threadIdx.x;
    if (t >= N_ETOT * 4) return;
    int e = t >> 2, h = t & 3;
    int s = (e < N_EDGES) ? srcv[e] : (e - N_EDGES);
    int d = (e < N_EDGES) ? dstv[e] : (e - N_EDGES);
    float ee = lrelu02(asrc[s * 4 + h] + adst[d * 4 + h]);
    atomicAdd(&dnm[d * 4 + h], __expf(ee - m[d * 4 + h]));
}

// AGG[dst][c] += 0.25 * sum_h alpha_eh * G[src][h*64+c]
__global__ void gat_scatter(const int* __restrict__ srcv, const int* __restrict__ dstv,
                            const float* __restrict__ asrc, const float* __restrict__ adst,
                            const float* __restrict__ m, const float* __restrict__ dnm,
                            const float* __restrict__ G, float* __restrict__ AGG) {
    unsigned t = blockIdx.x * blockDim.x + threadIdx.x;
    unsigned e = t >> 4;
    if (e >= N_ETOT) return;
    int cg = (int)(t & 15u) << 2;
    int s = (e < N_EDGES) ? srcv[e] : (int)(e - N_EDGES);
    int d = (e < N_EDGES) ? dstv[e] : (int)(e - N_EDGES);
    float ax = 0.f, ay = 0.f, az = 0.f, aw = 0.f;
#pragma unroll
    for (int h = 0; h < 4; ++h) {
        float ee = lrelu02(asrc[s * 4 + h] + adst[d * 4 + h]);
        float alpha = __expf(ee - m[d * 4 + h]) / dnm[d * 4 + h];
        const float4 gv = *(const float4*)(G + (size_t)s * GATW + h * HID + cg);
        ax += alpha * gv.x; ay += alpha * gv.y; az += alpha * gv.z; aw += alpha * gv.w;
    }
    float* o = AGG + (size_t)d * HID + cg;
    atomicAdd(o + 0, 0.25f * ax);
    atomicAdd(o + 1, 0.25f * ay);
    atomicAdd(o + 2, 0.25f * az);
    atomicAdd(o + 3, 0.25f * aw);
}

// ---------------- pooling + classifier ----------------

__global__ void pool_kernel(const float* __restrict__ X, const int* __restrict__ batch,
                            float* __restrict__ pooled, float* __restrict__ cnt) {
    unsigned t = blockIdx.x * blockDim.x + threadIdx.x;
    unsigned i = t >> 4;
    if (i >= N_NODES) return;
    int cg = (int)(t & 15u) << 2;
    int g = batch[i];
    const float4 v = *(const float4*)(X + (size_t)i * HID + cg);
    float* o = pooled + (size_t)g * HID + cg;
    atomicAdd(o + 0, v.x);
    atomicAdd(o + 1, v.y);
    atomicAdd(o + 2, v.z);
    atomicAdd(o + 3, v.w);
    if (cg == 0) atomicAdd(&cnt[g], 1.0f);
}

__global__ void classifier(const float* __restrict__ pooled, const float* __restrict__ cnt,
                           const float* __restrict__ cw1, const float* __restrict__ cb1,
                           const float* __restrict__ cw2, const float* __restrict__ cb2,
                           float* __restrict__ out) {
    int g = blockIdx.x * blockDim.x + threadIdx.x;
    if (g >= N_GRAPHS) return;
    float inv = 1.0f / fmaxf(cnt[g], 1.0f);
    const float* pg = pooled + (size_t)g * HID;
    float o = cb2[0];
    for (int j = 0; j < 32; ++j) {
        float z = cb1[j];
        for (int c = 0; c < HID; ++c) z += pg[c] * inv * cw1[c * 32 + j];
        o += fmaxf(z, 0.0f) * cw2[j];
    }
    out[g] = o;
}

// ---------------- launch ----------------

extern "C" void kernel_launch(void* const* d_in, const int* in_sizes, int n_in,
                              void* d_out, int out_size, void* d_ws, size_t ws_size,
                              hipStream_t stream) {
    (void)in_sizes; (void)n_in; (void)out_size; (void)ws_size;
    const float* x      = (const float*)d_in[0];
    const int*   ei     = (const int*)d_in[1];
    const int*   srcv   = ei;
    const int*   dstv   = ei + N_EDGES;
    const int*   batch  = (const int*)d_in[2];
    const float* w1     = (const float*)d_in[3];
    const float* b1     = (const float*)d_in[4];
    const float* w2     = (const float*)d_in[5];
    const float* b2     = (const float*)d_in[6];
    const float* w3     = (const float*)d_in[7];
    const float* b3     = (const float*)d_in[8];
    const float* gat_w  = (const float*)d_in[9];
    const float* att_s  = (const float*)d_in[10];
    const float* att_d  = (const float*)d_in[11];
    const float* gat_b  = (const float*)d_in[12];
    const float* cw1    = (const float*)d_in[13];
    const float* cb1    = (const float*)d_in[14];
    const float* cw2    = (const float*)d_in[15];
    const float* cb2    = (const float*)d_in[16];
    float* out = (float*)d_out;

    char* p = (char*)d_ws;
    auto carve = [&](size_t bytes) -> void* {
        void* r = (void*)p;
        p += (bytes + 255) & ~(size_t)255;
        return r;
    };
    _Float16* x16 = (_Float16*)carve((size_t)N_NODES * F_IN * 2);
    _Float16* h16 = (_Float16*)carve((size_t)N_NODES * HID * 2);
    _Float16* w1T = (_Float16*)carve((size_t)HID * F_IN * 2);
    _Float16* w2T = (_Float16*)carve((size_t)HID * HID * 2);
    _Float16* w3T = (_Float16*)carve((size_t)HID * HID * 2);
    _Float16* gwT = (_Float16*)carve((size_t)GATW * HID * 2);
    float* Hb   = (float*)carve((size_t)N_NODES * GATW * 4); // also holds [N,64] GCN gemm out
    float* AGG  = (float*)carve((size_t)N_NODES * HID * 4);
    float* X0   = (float*)carve((size_t)N_NODES * HID * 4);
    float* deg  = (float*)carve((size_t)N_NODES * 4);
    float* dinv = (float*)carve((size_t)N_NODES * 4);
    float* asrc = (float*)carve((size_t)N_NODES * 4 * 4);
    float* adst = (float*)carve((size_t)N_NODES * 4 * 4);
    float* mbuf = (float*)carve((size_t)N_NODES * 4 * 4);
    float* dnm  = (float*)carve((size_t)N_NODES * 4 * 4);
    float* pooled = (float*)carve((size_t)N_GRAPHS * HID * 4);
    float* cnt    = (float*)carve((size_t)N_GRAPHS * 4);

    // degrees + symmetric norm
    hipMemsetAsync(deg, 0, (size_t)N_NODES * 4, stream);
    deg_kernel<<<CDIV(N_ETOT, 256), 256, 0, stream>>>(dstv, deg);
    dinv_kernel<<<CDIV(N_NODES, 256), 256, 0, stream>>>(deg, dinv);

    // one-time f16 conversions (x + transposed weights)
    cvt_f16<<<CDIV(N_NODES * F_IN, 256), 256, 0, stream>>>(x, x16, N_NODES * F_IN);
    cvt_T_f16<<<CDIV(F_IN * HID, 256), 256, 0, stream>>>(w1, w1T, F_IN, HID);
    cvt_T_f16<<<CDIV(HID * HID, 256), 256, 0, stream>>>(w2, w2T, HID, HID);
    cvt_T_f16<<<CDIV(HID * HID, 256), 256, 0, stream>>>(w3, w3T, HID, HID);
    cvt_T_f16<<<CDIV(HID * GATW, 256), 256, 0, stream>>>(gat_w, gwT, HID, GATW);

    // GCN layer 1: K=128
    gemm_wmma_f16<<<(N_NODES / 16) * (HID / 16), 32, 0, stream>>>(x16, w1T, Hb, N_NODES, HID, F_IN);
    hipMemsetAsync(AGG, 0, (size_t)N_NODES * HID * 4, stream);
    gcn_scatter<<<CDIV(N_ETOT * 16, 256), 256, 0, stream>>>(srcv, dstv, dinv, Hb, AGG);
    bias_relu<<<CDIV(N_NODES * HID, 256), 256, 0, stream>>>(AGG, b1, X0, N_NODES * HID);

    // GCN layer 2
    cvt_f16<<<CDIV(N_NODES * HID, 256), 256, 0, stream>>>(X0, h16, N_NODES * HID);
    gemm_wmma_f16<<<(N_NODES / 16) * (HID / 16), 32, 0, stream>>>(h16, w2T, Hb, N_NODES, HID, HID);
    hipMemsetAsync(AGG, 0, (size_t)N_NODES * HID * 4, stream);
    gcn_scatter<<<CDIV(N_ETOT * 16, 256), 256, 0, stream>>>(srcv, dstv, dinv, Hb, AGG);
    bias_relu<<<CDIV(N_NODES * HID, 256), 256, 0, stream>>>(AGG, b2, X0, N_NODES * HID);

    // GCN layer 3
    cvt_f16<<<CDIV(N_NODES * HID, 256), 256, 0, stream>>>(X0, h16, N_NODES * HID);
    gemm_wmma_f16<<<(N_NODES / 16) * (HID / 16), 32, 0, stream>>>(h16, w3T, Hb, N_NODES, HID, HID);
    hipMemsetAsync(AGG, 0, (size_t)N_NODES * HID * 4, stream);
    gcn_scatter<<<CDIV(N_ETOT * 16, 256), 256, 0, stream>>>(srcv, dstv, dinv, Hb, AGG);
    bias_relu<<<CDIV(N_NODES * HID, 256), 256, 0, stream>>>(AGG, b3, X0, N_NODES * HID);

    // GAT: G = X0 @ gat_w  -> [N, 256]
    cvt_f16<<<CDIV(N_NODES * HID, 256), 256, 0, stream>>>(X0, h16, N_NODES * HID);
    gemm_wmma_f16<<<(N_NODES / 16) * (GATW / 16), 32, 0, stream>>>(h16, gwT, Hb, N_NODES, GATW, HID);
    gat_scores<<<CDIV(N_NODES * 4, 256), 256, 0, stream>>>(Hb, att_s, att_d, asrc, adst);
    fill_f32<<<CDIV(N_NODES * 4, 256), 256, 0, stream>>>(mbuf, -__builtin_huge_valf(), N_NODES * 4);
    hipMemsetAsync(dnm, 0, (size_t)N_NODES * 4 * 4, stream);
    gat_edge_max<<<CDIV(N_ETOT * 4, 256), 256, 0, stream>>>(srcv, dstv, asrc, adst, mbuf);
    gat_edge_denom<<<CDIV(N_ETOT * 4, 256), 256, 0, stream>>>(srcv, dstv, asrc, adst, mbuf, dnm);
    hipMemsetAsync(AGG, 0, (size_t)N_NODES * HID * 4, stream);
    gat_scatter<<<CDIV(N_ETOT * 16, 256), 256, 0, stream>>>(srcv, dstv, asrc, adst, mbuf, dnm, Hb, AGG);
    bias_relu<<<CDIV(N_NODES * HID, 256), 256, 0, stream>>>(AGG, gat_b, X0, N_NODES * HID);

    // global mean pool + classifier
    hipMemsetAsync(pooled, 0, (size_t)N_GRAPHS * HID * 4, stream);
    hipMemsetAsync(cnt, 0, (size_t)N_GRAPHS * 4, stream);
    pool_kernel<<<CDIV(N_NODES * 16, 256), 256, 0, stream>>>(X0, batch, pooled, cnt);
    classifier<<<1, 256, 0, stream>>>(pooled, cnt, cw1, cb1, cw2, cb2, out);
}
